// GConvLSTMWrapper_21680994910529
// MI455X (gfx1250) — matrix operand order repaired and checked by reading, not compiled, not run on hardware
//
#include <hip/hip_runtime.h>
#include <math.h>

typedef __attribute__((ext_vector_type(16))) __bf16 v16bf;
typedef __attribute__((ext_vector_type(8)))  float  v8f;

#define CIN 16
#define HIDDEN 64

// CDNA5 has a native V_TANH_F32 transcendental; use it if the toolchain exposes it.
#if __has_builtin(__builtin_amdgcn_tanhf)
__device__ __forceinline__ float fast_tanh(float v) { return __builtin_amdgcn_tanhf(v); }
#elif __has_builtin(__builtin_amdgcn_tanh_f32)
__device__ __forceinline__ float fast_tanh(float v) { return __builtin_amdgcn_tanh_f32(v); }
#else
__device__ __forceinline__ float fast_tanh(float v) { return tanhf(v); }
#endif

// sigmoid(v) == 0.5 + 0.5*tanh(v/2)  -> single v_tanh_f32 + fma
__device__ __forceinline__ float fast_sigmoid(float v) {
    return __builtin_fmaf(fast_tanh(0.5f * v), 0.5f, 0.5f);
}

// ---------------- scratch init ----------------
__global__ void k_zero(float* __restrict__ p, long n) {
    long i = (long)blockIdx.x * blockDim.x + threadIdx.x;
    if (i < n) p[i] = 0.0f;
}

// ---------------- deg[row] += w[e] ----------------
__global__ void k_degree(const long long* __restrict__ ei,
                         const float* __restrict__ w,
                         float* __restrict__ deg, int E) {
    int e = blockIdx.x * blockDim.x + threadIdx.x;
    if (e < E) {
        int r = (int)ei[e];
        __hip_atomic_fetch_add(&deg[r], w[e], __ATOMIC_RELAXED, __HIP_MEMORY_SCOPE_AGENT);
    }
}

// ---------------- dinv = deg>0 ? rsqrt(deg) : 0 ----------------
__global__ void k_dinv(const float* __restrict__ deg,
                       float* __restrict__ dinv, int n) {
    int i = blockIdx.x * blockDim.x + threadIdx.x;
    if (i < n) {
        float d = deg[i];
        dinv[i] = (d > 0.0f) ? rsqrtf(d) : 0.0f;
    }
}

// ---------------- norm[e] = -(dinv[row]*w*dinv[col]) ----------------
__global__ void k_norm(const long long* __restrict__ ei,
                       const float* __restrict__ w,
                       const float* __restrict__ dinv,
                       float* __restrict__ nrm, int E) {
    int e = blockIdx.x * blockDim.x + threadIdx.x;
    if (e < E) {
        int r = (int)ei[e];
        int c = (int)ei[(long)E + e];
        nrm[e] = -(dinv[r] * w[e] * dinv[c]);
    }
}

// ---------------- tx1[row,f] += norm[e]*x[col,f]  (thread per edge-feature) ----------------
__global__ void k_scatter(const long long* __restrict__ ei,
                          const float* __restrict__ nrm,
                          const float* __restrict__ x,
                          float* __restrict__ tx1, int E) {
    long tid = (long)blockIdx.x * blockDim.x + threadIdx.x;
    if (tid < (long)E * CIN) {
        int e = (int)(tid >> 4);
        int f = (int)(tid & 15);
        int r = (int)ei[e];
        int c = (int)ei[(long)E + e];
        float v = nrm[e] * x[c * CIN + f];
        __hip_atomic_fetch_add(&tx1[r * CIN + f], v, __ATOMIC_RELAXED, __HIP_MEMORY_SCOPE_AGENT);
    }
}

// ---------------- fused gates (WMMA bf16) + peephole LSTM + final linear ----------------
// Per wave: one 16-node M-tile. A = [x | tx1] (16x32 bf16), B = Wx for gates {i,c,o}
// as 12 fragments of 32x16 held in registers. D accumulated in f32.
__global__ __launch_bounds__(256, 1)
void k_gates(const float* __restrict__ x,   const float* __restrict__ tx1,
             const float* __restrict__ Wx,  const float* __restrict__ bx,
             const float* __restrict__ bh,  const float* __restrict__ wc,
             const float* __restrict__ bg,  const float* __restrict__ Wlin,
             const float* __restrict__ blin, float* __restrict__ out, int nN) {
    const int lane = threadIdx.x & 31;
    const int wib  = threadIdx.x >> 5;
    const int wpb  = blockDim.x >> 5;
    const int wave = blockIdx.x * wpb + wib;
    const int half = lane >> 4;     // which 16-lane half of the wave
    const int nl   = lane & 15;     // N (output col) within subtile; also M for A rows

    // ---- B fragments: gates i(0), c(2), o(3); 4 N-subtiles of 16 cols each ----
    // B 32x16 bf16, contiguous K order as in A: lanes 0-15 hold K=0..15, lanes 16-31 hold K=16..31,
    // element e of the 16-elt vector is K = half*16 + e at column nl.
    // W_cat row k: k<16 -> Wx[g][0][k][col], k>=16 -> Wx[g][1][k-16][col]
    //  => element e = Wx[((g*2 + half)*16 + e)*64 + col]
    v16bf Bf[3][4];
    float bsum[3][4], wc2[4], wl[4];
    const int gsel[3] = {0, 2, 3};
#pragma unroll
    for (int gi = 0; gi < 3; ++gi) {
        const int g = gsel[gi];
#pragma unroll
        for (int nt = 0; nt < 4; ++nt) {
            const int col = nt * 16 + nl;
            const float* wp = Wx + (size_t)((g * 2 + half) * 16) * HIDDEN + col;
            v16bf b;
#pragma unroll
            for (int e = 0; e < 16; ++e) b[e] = (__bf16)wp[(size_t)e * HIDDEN];
            Bf[gi][nt] = b;
            bsum[gi][nt] = bx[g * HIDDEN + col] + bh[g * HIDDEN + col] + bg[g * HIDDEN + col];
        }
    }
#pragma unroll
    for (int nt = 0; nt < 4; ++nt) {
        const int col = nt * 16 + nl;
        wc2[nt] = wc[2 * HIDDEN + col];
        wl[nt]  = Wlin[col];
    }
    const float b0 = blin[0];

    const int nTiles = (nN + 15) >> 4;
    const int stride = gridDim.x * wpb;
    for (int tile = wave; tile < nTiles; tile += stride) {
        const int base = tile * 16;
        int node = base + nl;
        if (node >= nN) node = nN - 1;   // N%16==0 in practice; safe clamp

        // ---- A 16x32 bf16, contiguous K order: elem e -> K = half*16 + e; row M = nl ----
        const float* xp = (half ? tx1 : x) + (size_t)node * CIN;
        v16bf A;
#pragma unroll
        for (int e = 0; e < 16; ++e) A[e] = (__bf16)xp[e];

        v8f acc_i[4], acc_c[4], acc_o[4];
        const v8f z = {0.f, 0.f, 0.f, 0.f, 0.f, 0.f, 0.f, 0.f};
#pragma unroll
        for (int nt = 0; nt < 4; ++nt) {
            acc_i[nt] = __builtin_amdgcn_wmma_f32_16x16x32_bf16(false, A, false, Bf[0][nt],
                                                                (short)0, z, false, false);
            acc_c[nt] = __builtin_amdgcn_wmma_f32_16x16x32_bf16(false, A, false, Bf[1][nt],
                                                                (short)0, z, false, false);
            acc_o[nt] = __builtin_amdgcn_wmma_f32_16x16x32_bf16(false, A, false, Bf[2][nt],
                                                                (short)0, z, false, false);
        }

        // ---- nonlinearities in C-fragment layout: VGPR r -> M = half*8 + r, col = nt*16+nl ----
        // All four nonlinearities expressed via native v_tanh_f32 (sigmoid(v)=0.5+0.5*tanh(v/2)).
        float res[8];
#pragma unroll
        for (int r = 0; r < 8; ++r) res[r] = 0.f;
#pragma unroll
        for (int nt = 0; nt < 4; ++nt) {
#pragma unroll
            for (int r = 0; r < 8; ++r) {
                float iv = fast_sigmoid(acc_i[nt][r] + bsum[0][nt]);
                float tv = fast_tanh(acc_c[nt][r] + bsum[1][nt]);
                float cv = iv * tv;                         // C = I*T (F*C0 == 0)
                float ov = fast_sigmoid(acc_o[nt][r] + bsum[2][nt] + wc2[nt] * cv);
                float hv = ov * fast_tanh(cv);
                res[r] += hv * wl[nt];                      // partial H @ Wlin
            }
        }

        // ---- reduce across the 16 lanes of each half (sum over output columns) ----
#pragma unroll
        for (int r = 0; r < 8; ++r) {
            float s = res[r];
            s += __shfl_xor(s, 1, 32);
            s += __shfl_xor(s, 2, 32);
            s += __shfl_xor(s, 4, 32);
            s += __shfl_xor(s, 8, 32);
            const int m = half * 8 + r;
            if (nl == 0 && base + m < nN) out[base + m] = s + b0;
        }
    }
}

extern "C" void kernel_launch(void* const* d_in, const int* in_sizes, int n_in,
                              void* d_out, int out_size, void* d_ws, size_t ws_size,
                              hipStream_t stream) {
    const float*     x    = (const float*)d_in[0];
    const long long* ei   = (const long long*)d_in[1];   // int64 edge_index [2,E]
    const float*     ew   = (const float*)d_in[2];
    const float*     Wx   = (const float*)d_in[3];       // [4,2,16,64]
    const float*     bx   = (const float*)d_in[4];       // [4,64]
    // d_in[5] = Wh: dead (initial H == 0)
    const float*     bh   = (const float*)d_in[6];       // [4,64]
    const float*     wc   = (const float*)d_in[7];       // [3,64]
    const float*     bg   = (const float*)d_in[8];       // [4,64]
    const float*     Wlin = (const float*)d_in[9];       // [64,1]
    const float*     blin = (const float*)d_in[10];      // [1]
    float*           out  = (float*)d_out;

    const int N = in_sizes[0] / CIN;
    const int E = in_sizes[2];

    // scratch layout: deg[N] | tx1[16N] (both zeroed together) | dinv[N] | norm[E]
    float* deg  = (float*)d_ws;
    float* tx1  = deg + N;
    float* dinv = tx1 + (size_t)16 * N;
    float* nrm  = dinv + N;

    {
        long tot = 17L * N;
        int blocks = (int)((tot + 255) / 256);
        k_zero<<<blocks, 256, 0, stream>>>(deg, tot);
    }
    k_degree<<<(E + 255) / 256, 256, 0, stream>>>(ei, ew, deg, E);
    k_dinv<<<(N + 255) / 256, 256, 0, stream>>>(deg, dinv, N);
    k_norm<<<(E + 255) / 256, 256, 0, stream>>>(ei, ew, dinv, nrm, E);
    {
        long tot = (long)E * CIN;
        int blocks = (int)((tot + 255) / 256);
        k_scatter<<<blocks, 256, 0, stream>>>(ei, nrm, x, tx1, E);
    }
    {
        const int nTiles = (N + 15) / 16;
        const int wpb = 8;                              // 256 threads = 8 waves
        const int blocks = (nTiles + wpb - 1) / wpb;
        k_gates<<<blocks, 256, 0, stream>>>(x, tx1, Wx, bx, bh, wc, bg, Wlin, blin, out, N);
    }
}
